// STAttention_notf_v7_11690900980418
// MI455X (gfx1250) — matrix operand order repaired and checked by reading, not compiled, not run on hardware
//
#include <hip/hip_runtime.h>
#include <math.h>

// ---------------------------------------------------------------------------
// STAttention RNN for MI455X (gfx1250): bf16 WMMA GEMMs + fused epilogues.
// All matrix work goes through v_wmma_f32_16x16x32_bf16. Weights are converted
// to bf16 once (resident in 192MB L2); activations stay fp32 and are converted
// to bf16 in-register (packed cvt) when loaded as the WMMA A operand.
// Workspace requirement: ~101 MB.
// ---------------------------------------------------------------------------

#define B_     256
#define H_     512
#define DIN_   512
#define LENC_  48
#define LOOK_  10
#define LT_    58      // LENC+LDEC
#define START_ 48      // LDEC + LENC - LOOK
#define FH_    2048    // 4*H
#define BH_    (B_*H_) // 131072

typedef __bf16 bf16_t;
typedef __attribute__((ext_vector_type(16))) __bf16 v16bf;
typedef __attribute__((ext_vector_type(8)))  __bf16 v8bf;
typedef __attribute__((ext_vector_type(8)))  float  v8f;

static __device__ __forceinline__ bf16_t f2bf(float f) {
  unsigned u = __builtin_bit_cast(unsigned, f);
  unsigned r = u + 0x7FFFu + ((u >> 16) & 1u);   // round-to-nearest-even
  unsigned short s = (unsigned short)(r >> 16);
  return __builtin_bit_cast(bf16_t, s);
}
static __device__ __forceinline__ float sigm_(float x) { return 1.0f / (1.0f + __expf(-x)); }

// One K-step (K advances by 32): load 16x32 bf16 A tile from an fp32 row,
// 4x (32x16) bf16 B tiles from the weight matrix, and issue 4 WMMAs.
// aRowK = &A[mRow, kk] for this lane's row; W row stride = K.
static __device__ __forceinline__ void gemm_step(const float* __restrict__ aRowK,
                                                 const bf16_t* __restrict__ W,
                                                 int K, int kk, int half, int l16,
                                                 int nBase, v8f acc[4]) {
  // A lane layout (matches B): lanes 0-15 hold K 0..15;
  // lanes 16-31 hold K 16..31.
  v8f fa = *(const v8f*)(aRowK + half * 16);
  v8f fb = *(const v8f*)(aRowK + half * 16 + 8);
  v8bf lo = __builtin_convertvector(fa, v8bf);
  v8bf hi = __builtin_convertvector(fb, v8bf);
  v16bf a = __builtin_shufflevector(lo, hi, 0, 1, 2, 3, 4, 5, 6, 7,
                                    8, 9, 10, 11, 12, 13, 14, 15);
#pragma unroll
  for (int t = 0; t < 4; ++t) {
    // B lane layout: lane n reads 16 contiguous bf16 of W row n,
    // lanes 0-15 -> K=kk..kk+15, lanes 16-31 -> K=kk+16..kk+31.
    const int n = nBase + t * 16 + l16;
    v16bf b = *(const v16bf*)(W + (size_t)n * K + kk + half * 16);
    acc[t] = __builtin_amdgcn_wmma_f32_16x16x32_bf16(
        false, a, false, b, (short)0, acc[t], false, false);
  }
}

// ---------------------------------------------------------------------------
// GEMM: C[M,N] = concat(A0,A1)[M,K] @ W[N,K]^T  (+ epilogue)
// EPI 0: + bias[n] (bias may be null)
// EPI 1: tanh(acc + add0[m,n] + add1[m,n])
// EPI 2: + bias[n] + lab[m]*w0[n]   (target-LSTM first-column fold-in)
// grid = (M/16, N/256), block = 128 (4 waves, each wave -> 16x64 strip)
// K0 and K are multiples of 32, so every 32-wide K-step lives entirely in one
// of A0/A1 -> the source split is scalar (no per-lane address selects).
// ---------------------------------------------------------------------------
template <int EPI>
__global__ __launch_bounds__(128)
void gemm_wmma(const float* __restrict__ A0, int lda0, int K0,
               const float* __restrict__ A1, int lda1,
               const bf16_t* __restrict__ W,
               float* __restrict__ C, int ldc,
               int N, int K,
               const float* __restrict__ bias,
               const float* __restrict__ add0, int ldadd0,
               const float* __restrict__ add1, int ldadd1,
               const float* __restrict__ lab, const float* __restrict__ w0) {
  const int lane = threadIdx.x & 31;
  const int wave = threadIdx.x >> 5;
  const int half = lane >> 4;     // 0: lanes 0-15, 1: lanes 16-31
  const int l16  = lane & 15;
  const int mRow = blockIdx.x * 16 + l16;        // A row fetched by this lane
  const int nBase = blockIdx.y * 256 + wave * 64;

  v8f acc[4] = {v8f{}, v8f{}, v8f{}, v8f{}};

  const float* aRow0 = A0 + (size_t)mRow * lda0;
  int kk = 0;
  for (; kk < K0; kk += 32) {
    // rotate the prefetched N-tile row so all 4 tiles get near-cache coverage
    if (kk + 64 < K) {
      const int pn = nBase + 16 * ((kk >> 5) & 3) + l16;
      __builtin_prefetch((const void*)(W + (size_t)pn * K + kk + 64), 0, 3);
    }
    gemm_step(aRow0 + kk, W, K, kk, half, l16, nBase, acc);
  }
  if (kk < K) {
    const float* aRow1 = A1 + (size_t)mRow * lda1;
    for (; kk < K; kk += 32) {
      if (kk + 64 < K) {
        const int pn = nBase + 16 * ((kk >> 5) & 3) + l16;
        __builtin_prefetch((const void*)(W + (size_t)pn * K + kk + 64), 0, 3);
      }
      gemm_step(aRow1 + (kk - K0), W, K, kk, half, l16, nBase, acc);
    }
  }

  // C/D layout: element r -> row mBase+r, col nBase + t*16 + (lane&15)
  const int mBase = blockIdx.x * 16 + half * 8;
#pragma unroll
  for (int t = 0; t < 4; ++t) {
    const int n = nBase + t * 16 + l16;
#pragma unroll
    for (int r = 0; r < 8; ++r) {
      const int m = mBase + r;
      float v = acc[t][r];
      if (EPI == 0) { if (bias) v += bias[n]; }
      else if (EPI == 1) {
        v = tanhf(v + add0[(size_t)m * ldadd0 + n] + add1[(size_t)m * ldadd1 + n]);
      } else if (EPI == 2) {
        v += bias[n] + lab[m] * w0[n];
      }
      C[(size_t)m * ldc + n] = v;
    }
  }
}

// --------------------------- elementwise kernels ---------------------------

__global__ __launch_bounds__(256)
void lstm_ew(const float* __restrict__ G, const float* __restrict__ c_in,
             float* __restrict__ c_out, float* __restrict__ h0, float* __restrict__ h1) {
  const int idx = blockIdx.x * 256 + threadIdx.x;  // over B*H
  const int b = idx >> 9, h = idx & 511;
  const float* g = G + (size_t)b * FH_;
  const float i = sigm_(g[h]);
  const float f = sigm_(g[512 + h]);
  const float gg = tanhf(g[1024 + h]);
  const float o = sigm_(g[1536 + h]);
  const float c2 = f * c_in[idx] + i * gg;
  const float hh = o * tanhf(c2);
  c_out[idx] = c2;
  h0[idx] = hh;
  if (h1) h1[idx] = hh;
}

__global__ __launch_bounds__(256)
void softmax_mul(const float* __restrict__ score, const float* __restrict__ xin,
                 long ldx, float* __restrict__ xmod) {
  __shared__ float sm[256];
  const int b = blockIdx.x, tid = threadIdx.x;
  const float* s = score + (size_t)b * 512;
  const float v0 = s[tid], v1 = s[tid + 256];
  sm[tid] = fmaxf(v0, v1);
  __syncthreads();
  for (int o = 128; o > 0; o >>= 1) { if (tid < o) sm[tid] = fmaxf(sm[tid], sm[tid + o]); __syncthreads(); }
  const float mx = sm[0];
  __syncthreads();
  const float e0 = __expf(v0 - mx), e1 = __expf(v1 - mx);
  sm[tid] = e0 + e1;
  __syncthreads();
  for (int o = 128; o > 0; o >>= 1) { if (tid < o) sm[tid] += sm[tid + o]; __syncthreads(); }
  const float inv = 1.0f / sm[0];
  const float* x = xin + (size_t)b * ldx;
  xmod[(size_t)b * 512 + tid]       = x[tid] * e0 * inv;
  xmod[(size_t)b * 512 + tid + 256] = x[tid + 256] * e1 * inv;
}

__global__ __launch_bounds__(256)
void target_dot(const float* __restrict__ ht, const float* __restrict__ regw,
                const float* __restrict__ regb, float* __restrict__ out,
                float* __restrict__ lab, int t) {
  __shared__ float sm[256];
  const int b = blockIdx.x, tid = threadIdx.x;
  sm[tid] = ht[(size_t)b * 512 + tid] * regw[tid] +
            ht[(size_t)b * 512 + 256 + tid] * regw[tid + 256];
  __syncthreads();
  for (int o = 128; o > 0; o >>= 1) { if (tid < o) sm[tid] += sm[tid + o]; __syncthreads(); }
  if (tid == 0) {
    const float v = sm[0] + regb[0];
    out[(size_t)b * LOOK_ + t] = v;
    lab[b] = v;
  }
}

__global__ __launch_bounds__(128)
void score_dec(const float* __restrict__ wh, const float* __restrict__ U,
               const float* __restrict__ Vw, const float* __restrict__ Vb,
               float* __restrict__ sc) {
  __shared__ float sm[128];
  const int r = blockIdx.x, tid = threadIdx.x;  // r = l*B + b
  const int b = r & (B_ - 1);
  float p = 0.f;
  for (int j = tid; j < 512; j += 128)
    p += tanhf(wh[(size_t)b * 512 + j] + U[(size_t)r * 512 + j]) * Vw[j];
  sm[tid] = p;
  __syncthreads();
  for (int o = 64; o > 0; o >>= 1) { if (tid < o) sm[tid] += sm[tid + o]; __syncthreads(); }
  if (tid == 0) sc[r] = sm[0] + Vb[0];
}

__global__ __launch_bounds__(256)
void decin_k(const float* __restrict__ sc, const float* __restrict__ mid2,
             float* __restrict__ dec) {
  const int idx = blockIdx.x * 256 + threadIdx.x;  // B*H
  const int b = idx >> 9, h = idx & 511;
  float s = 0.f;
  for (int l = 0; l < LENC_; ++l)
    s += sc[l * B_ + b] * mid2[((size_t)(l * B_ + b)) * 512 + h];
  dec[idx] = s;
}

// --------------------------- prep kernels ---------------------------

__global__ __launch_bounds__(256)
void convcat_k(const float* __restrict__ S0, int ld0, int off0, int K0,
               const float* __restrict__ S1, int ld1, int off1, int K1,
               bf16_t* __restrict__ D, int total) {
  const int i = blockIdx.x * 256 + threadIdx.x;
  if (i >= total) return;
  const int K = K0 + K1;
  const int n = i / K, k = i - n * K;
  const float v = (k < K0) ? S0[(size_t)n * ld0 + off0 + k]
                           : S1[(size_t)n * ld1 + off1 + (k - K0)];
  D[i] = f2bf(v);
}

__global__ __launch_bounds__(256)
void vec_add_k(const float* __restrict__ a, const float* __restrict__ b,
               float* __restrict__ o, int n) {
  const int i = blockIdx.x * 256 + threadIdx.x;
  if (i < n) o[i] = a[i] + b[i];
}

__global__ __launch_bounds__(256)
void gather_k(const float* __restrict__ s, int stride, int off,
              float* __restrict__ d, int n) {
  const int i = blockIdx.x * 256 + threadIdx.x;
  if (i < n) d[i] = s[(size_t)i * stride + off];
}

__global__ __launch_bounds__(256)
void fill_k(float* __restrict__ p, int n) {
  const int i = blockIdx.x * 256 + threadIdx.x;
  if (i < n) p[i] = 0.f;
}

// --------------------------- host-side launch helper ---------------------------

static void launch_gemm(int epi, const float* A0, int lda0, int K0,
                        const float* A1, int lda1, const bf16_t* W,
                        float* C, int ldc, int M, int N, int K,
                        const float* bias, const float* add0, int la0,
                        const float* add1, int la1, const float* lab,
                        const float* w0, hipStream_t stream) {
  dim3 g(M / 16, N / 256), b(128);
  if (epi == 0)
    gemm_wmma<0><<<g, b, 0, stream>>>(A0, lda0, K0, A1, lda1, W, C, ldc, N, K, bias, add0, la0, add1, la1, lab, w0);
  else if (epi == 1)
    gemm_wmma<1><<<g, b, 0, stream>>>(A0, lda0, K0, A1, lda1, W, C, ldc, N, K, bias, add0, la0, add1, la1, lab, w0);
  else
    gemm_wmma<2><<<g, b, 0, stream>>>(A0, lda0, K0, A1, lda1, W, C, ldc, N, K, bias, add0, la0, add1, la1, lab, w0);
}

extern "C" void kernel_launch(void* const* d_in, const int* in_sizes, int n_in,
                              void* d_out, int out_size, void* d_ws, size_t ws_size,
                              hipStream_t stream) {
  (void)in_sizes; (void)n_in; (void)out_size; (void)ws_size;

  const float* input   = (const float*)d_in[0];   // (B, LENC, DIN)
  const float* label_p = (const float*)d_in[1];   // (B, 58)
  const float* Wih_sp  = (const float*)d_in[2];
  const float* Whh_sp  = (const float*)d_in[3];
  const float* bih_sp  = (const float*)d_in[4];
  const float* bhh_sp  = (const float*)d_in[5];
  const float* Wih_tg  = (const float*)d_in[6];   // (4H, 513)
  const float* Whh_tg  = (const float*)d_in[7];
  const float* bih_tg  = (const float*)d_in[8];
  const float* bhh_tg  = (const float*)d_in[9];
  const float* Wih_mid = (const float*)d_in[10];
  const float* Whh_mid = (const float*)d_in[11];
  const float* bih_mid = (const float*)d_in[12];
  const float* bhh_mid = (const float*)d_in[13];
  const float* Wi_w = (const float*)d_in[14];
  const float* Wi_b = (const float*)d_in[15];
  const float* We_w = (const float*)d_in[16];     // (H, 2H)
  const float* Wt_w = (const float*)d_in[17];     // (H, 2H)
  const float* Vd_w = (const float*)d_in[18];     // (DIN, H)
  const float* Vd_b = (const float*)d_in[19];
  const float* Wx_w = (const float*)d_in[20];     // (H, H)
  const float* Wx_b = (const float*)d_in[21];
  const float* Wh_w = (const float*)d_in[22];     // (H, 2H)
  const float* V_w  = (const float*)d_in[23];
  const float* V_b  = (const float*)d_in[24];
  const float* reg_w = (const float*)d_in[25];
  const float* reg_b = (const float*)d_in[26];
  float* out = (float*)d_out;

  // ---- workspace carve-up ----
  char* wsp = (char*)d_ws;
  auto alloc = [&](size_t bytes) -> char* {
    char* r = wsp; wsp += (bytes + 255) & ~(size_t)255; return r;
  };

  bf16_t* WspC  = (bf16_t*)alloc((size_t)FH_ * 1024 * 2);
  bf16_t* WmidC = (bf16_t*)alloc((size_t)FH_ * 1024 * 2);
  bf16_t* WtgC  = (bf16_t*)alloc((size_t)FH_ * 1024 * 2);
  bf16_t* WeB   = (bf16_t*)alloc((size_t)H_ * 1024 * 2);
  bf16_t* WtB   = (bf16_t*)alloc((size_t)H_ * 1024 * 2);
  bf16_t* WhB   = (bf16_t*)alloc((size_t)H_ * 1024 * 2);
  bf16_t* VdB   = (bf16_t*)alloc((size_t)DIN_ * H_ * 2);
  bf16_t* WxB   = (bf16_t*)alloc((size_t)H_ * H_ * 2);
  bf16_t* WiB   = (bf16_t*)alloc((size_t)H_ * DIN_ * 2);
  float* b_sp  = (float*)alloc(FH_ * 4);
  float* b_tg  = (float*)alloc(FH_ * 4);
  float* b_mid = (float*)alloc(FH_ * 4);
  float* w0tg  = (float*)alloc(FH_ * 4);
  float* lab   = (float*)alloc(B_ * 4);
  float* wi_seq = (float*)alloc((size_t)B_ * LENC_ * H_ * 4);   // rows r=b*48+l
  float* mid_buf = (float*)alloc((size_t)LENC_ * B_ * H_ * 4);  // rows r=l*B+b; reused as U
  float* mid2   = (float*)alloc((size_t)LENC_ * B_ * H_ * 4);
  float* G      = (float*)alloc((size_t)B_ * FH_ * 4);
  float* tmp1   = (float*)alloc((size_t)BH_ * 4);
  float* score  = (float*)alloc((size_t)BH_ * 4);
  float* xmod   = (float*)alloc((size_t)BH_ * 4);
  float* heB = (float*)alloc((size_t)4 * BH_ * 4);
  float* he = heB, *ce = heB + BH_, *hmid = heB + 2 * BH_, *cmid = heB + 3 * BH_;
  float* htB = (float*)alloc((size_t)3 * BH_ * 4);
  float* ht = htB, *ct = htB + BH_, *dec_in = htB + 2 * BH_;
  float* wt = (float*)alloc((size_t)BH_ * 4);
  float* wh = (float*)alloc((size_t)BH_ * 4);
  float* sc = (float*)alloc((size_t)LENC_ * B_ * 4);
  float* U = mid_buf;  // alias: mid_buf is dead when U is computed

  // ---- one-time prep ----
  {
    int tot = FH_ * 1024;
    convcat_k<<<(tot + 255) / 256, 256, 0, stream>>>(Wih_sp, 512, 0, 512, Whh_sp, 512, 0, 512, WspC, tot);
    convcat_k<<<(tot + 255) / 256, 256, 0, stream>>>(Wih_mid, 512, 0, 512, Whh_mid, 512, 0, 512, WmidC, tot);
    convcat_k<<<(tot + 255) / 256, 256, 0, stream>>>(Wih_tg, 513, 1, 512, Whh_tg, 512, 0, 512, WtgC, tot);
    tot = H_ * 1024;
    convcat_k<<<(tot + 255) / 256, 256, 0, stream>>>(We_w, 1024, 0, 1024, nullptr, 0, 0, 0, WeB, tot);
    convcat_k<<<(tot + 255) / 256, 256, 0, stream>>>(Wt_w, 1024, 0, 1024, nullptr, 0, 0, 0, WtB, tot);
    convcat_k<<<(tot + 255) / 256, 256, 0, stream>>>(Wh_w, 1024, 0, 1024, nullptr, 0, 0, 0, WhB, tot);
    tot = H_ * 512;
    convcat_k<<<(tot + 255) / 256, 256, 0, stream>>>(Vd_w, 512, 0, 512, nullptr, 0, 0, 0, VdB, tot);
    convcat_k<<<(tot + 255) / 256, 256, 0, stream>>>(Wx_w, 512, 0, 512, nullptr, 0, 0, 0, WxB, tot);
    convcat_k<<<(tot + 255) / 256, 256, 0, stream>>>(Wi_w, 512, 0, 512, nullptr, 0, 0, 0, WiB, tot);
    vec_add_k<<<FH_ / 256, 256, 0, stream>>>(bih_sp, bhh_sp, b_sp, FH_);
    vec_add_k<<<FH_ / 256, 256, 0, stream>>>(bih_tg, bhh_tg, b_tg, FH_);
    vec_add_k<<<FH_ / 256, 256, 0, stream>>>(bih_mid, bhh_mid, b_mid, FH_);
    gather_k<<<FH_ / 256, 256, 0, stream>>>(Wih_tg, 513, 0, w0tg, FH_);
    gather_k<<<1, 256, 0, stream>>>(label_p, LT_, START_, lab, B_);
    fill_k<<<(3 * BH_) / 256, 256, 0, stream>>>(htB, 3 * BH_);  // ht, ct, dec_in
    // wi_seq: (B*LENC, H) = input @ Wi^T + Wi_b
    launch_gemm(0, input, DIN_, DIN_, nullptr, 0, WiB, wi_seq, H_,
                B_ * LENC_, H_, DIN_, Wi_b, nullptr, 0, nullptr, 0, nullptr, nullptr, stream);
  }

  // ---- outer decode loop ----
  for (int t = 0; t < LOOK_; ++t) {
    // target LSTM: G = [dec_in|ht] @ WtgC^T + b_tg + lab*w0
    launch_gemm(2, dec_in, H_, H_, ht, H_, WtgC, G, FH_, B_, FH_, 1024,
                b_tg, nullptr, 0, nullptr, 0, lab, w0tg, stream);
    lstm_ew<<<BH_ / 256, 256, 0, stream>>>(G, ct, ct, ht, nullptr);
    target_dot<<<B_, 256, 0, stream>>>(ht, reg_w, reg_b, out, lab, t);
    // wt = [ht|ct] @ Wt^T
    launch_gemm(0, ht, H_, H_, ct, H_, WtB, wt, H_, B_, H_, 1024,
                nullptr, nullptr, 0, nullptr, 0, nullptr, nullptr, stream);

    fill_k<<<(4 * BH_) / 256, 256, 0, stream>>>(heB, 4 * BH_);  // he,ce,hmid,cmid

    // sp scan
    for (int l = 0; l < LENC_; ++l) {
      // tmp1 = tanh([he|ce]@We^T + wi_seq[:,l,:] + wt)
      launch_gemm(1, he, H_, H_, ce, H_, WeB, tmp1, H_, B_, H_, 1024,
                  nullptr, wi_seq + (size_t)l * H_, LENC_ * H_, wt, H_,
                  nullptr, nullptr, stream);
      // score = tmp1 @ Vd^T + Vd_b
      launch_gemm(0, tmp1, H_, H_, nullptr, 0, VdB, score, DIN_, B_, DIN_, H_,
                  Vd_b, nullptr, 0, nullptr, 0, nullptr, nullptr, stream);
      // x_mod = x_l * softmax(score)
      softmax_mul<<<B_, 256, 0, stream>>>(score, input + (size_t)l * DIN_,
                                          (long)LENC_ * DIN_, xmod);
      // G = [x_mod|he] @ WspC^T + b_sp
      launch_gemm(0, xmod, DIN_, DIN_, he, H_, WspC, G, FH_, B_, FH_, 1024,
                  b_sp, nullptr, 0, nullptr, 0, nullptr, nullptr, stream);
      lstm_ew<<<BH_ / 256, 256, 0, stream>>>(G, ce, ce, he, mid_buf + (size_t)l * BH_);
    }

    // mid scan
    for (int l = 0; l < LENC_; ++l) {
      launch_gemm(0, mid_buf + (size_t)l * BH_, H_, H_, hmid, H_, WmidC, G, FH_,
                  B_, FH_, 1024, b_mid, nullptr, 0, nullptr, 0, nullptr, nullptr, stream);
      lstm_ew<<<BH_ / 256, 256, 0, stream>>>(G, cmid, cmid, hmid, mid2 + (size_t)l * BH_);
    }

    // attention readout
    launch_gemm(0, ht, H_, H_, ct, H_, WhB, wh, H_, B_, H_, 1024,
                nullptr, nullptr, 0, nullptr, 0, nullptr, nullptr, stream);
    // U = mid2 @ Wx^T + Wx_b   (M = LENC*B; overwrites mid_buf, which is dead)
    launch_gemm(0, mid2, H_, H_, nullptr, 0, WxB, U, H_, LENC_ * B_, H_, H_,
                Wx_b, nullptr, 0, nullptr, 0, nullptr, nullptr, stream);
    score_dec<<<LENC_ * B_, 128, 0, stream>>>(wh, U, V_w, V_b, sc);
    decin_k<<<BH_ / 256, 256, 0, stream>>>(sc, mid2, dec_in);
  }
}
